// SelfModifyingTitans_34041910788567
// MI455X (gfx1250) — compile-verified, matrix operand order checked by reading, not run on hardware
//
#include <hip/hip_runtime.h>
#include <hip/hip_bf16.h>
#include <math.h>

typedef __bf16 bf16;
typedef __attribute__((ext_vector_type(16))) __bf16 v16bf;
typedef __attribute__((ext_vector_type(8)))  __bf16 v8bf;
typedef __attribute__((ext_vector_type(8)))  float  v8f;

#define Bb 4
#define Ss 2048
#define Dd 768
#define Hh 12
#define HDd 64
#define Pp 768
#define ROWS (Bb*Ss)      /* 8192 token rows */
#define DH  (Dd/4)        /* 192 modifier hidden */

// ---------------------------------------------------------------- utilities

__device__ __forceinline__ float wred32(float x) {
#pragma unroll
  for (int m = 1; m < 32; m <<= 1) x += __shfl_xor(x, m, 32);
  return x;
}

__global__ void cvt_bf16_kernel(const float* __restrict__ in, bf16* __restrict__ out, int n) {
  int i = blockIdx.x * blockDim.x + threadIdx.x;
  if (i < n) out[i] = (bf16)in[i];
}

// ------------------------------------------------- input LayerNorm -> bf16

__global__ void ln_in_kernel(const float* __restrict__ x, const float* __restrict__ g,
                             const float* __restrict__ b, bf16* __restrict__ xn_bf) {
  __shared__ float red[16];
  const int row = blockIdx.x;
  const float* xr = x + (size_t)row * Dd;
  const int wave = threadIdx.x >> 5, lane = threadIdx.x & 31;

  float s = 0.f;
  for (int i = threadIdx.x; i < Dd; i += 256) s += xr[i];
  s = wred32(s);
  if (lane == 0) red[wave] = s;
  __syncthreads();
  float mean = 0.f;
#pragma unroll
  for (int i = 0; i < 8; ++i) mean += red[i];
  mean *= (1.0f / Dd);

  float vs = 0.f;
  for (int i = threadIdx.x; i < Dd; i += 256) { float d = xr[i] - mean; vs += d * d; }
  vs = wred32(vs);
  if (lane == 0) red[8 + wave] = vs;
  __syncthreads();
  float var = 0.f;
#pragma unroll
  for (int i = 0; i < 8; ++i) var += red[8 + i];
  float rstd = rsqrtf(var * (1.0f / Dd) + 1e-5f);

  for (int i = threadIdx.x; i < Dd; i += 256) {
    float y = (xr[i] - mean) * rstd * g[i] + b[i];
    xn_bf[(size_t)row * Dd + i] = (bf16)y;
  }
}

// --------------------------------------------- per-head LayerNorm in place

__global__ void head_ln_kernel(float* __restrict__ data, const float* __restrict__ g,
                               const float* __restrict__ b) {
  const int wave = threadIdx.x >> 5, lane = threadIdx.x & 31;
  const int row = blockIdx.x * 8 + wave;                // row of 64 floats
  float* p = data + (size_t)row * HDd;
  float a0 = p[lane], a1 = p[lane + 32];
  float mean = wred32(a0 + a1) * (1.0f / HDd);
  float d0 = a0 - mean, d1 = a1 - mean;
  float var = wred32(d0 * d0 + d1 * d1) * (1.0f / HDd);
  float rstd = rsqrtf(var + 1e-5f);
  p[lane]      = d0 * rstd * g[lane]      + b[lane];
  p[lane + 32] = d1 * rstd * g[lane + 32] + b[lane + 32];
}

// ------------------------------------------------------- WMMA bf16 GEMMs
// A (M,K) row-major bf16 tokens; W (N,K) row-major bf16 (PyTorch Linear layout),
// computes A @ W^T. Per-lane fragment layout for 16x16x32 bf16 (ISA 7.12.2):
// lane = half*16 + r ; element j (0..15): K = (j<8?0:16) + half*8 + (j&7)
// -> two contiguous 8-element (16B) chunks at kk+half*8 and kk+16+half*8.

__device__ __forceinline__ v16bf load_frag(const bf16* __restrict__ base, int ld,
                                           int row, int kk, int half) {
  const bf16* p = base + (size_t)row * ld + kk + half * 8;
  v8bf lo = *(const v8bf*)(p);
  v8bf hi = *(const v8bf*)(p + 16);
  v16bf f;
#pragma unroll
  for (int j = 0; j < 8; ++j) { f[j] = lo[j]; f[j + 8] = hi[j]; }
  return f;
}

// out_bf = GELU(A @ W^T + bias), bf16 store. Wave: 64x16 tile, block: 512x16.
__global__ void gemm_gelu_kernel(const bf16* __restrict__ A, const bf16* __restrict__ W,
                                 const float* __restrict__ bias, bf16* __restrict__ out,
                                 int N, int K) {
  const int wv = threadIdx.x >> 5;
  const int lane = threadIdx.x & 31;
  const int half = lane >> 4;
  const int r16 = lane & 15;
  const int n0 = blockIdx.x * 16;
  const int m0 = blockIdx.y * 512 + wv * 64;

  v8f acc[4] = {};
  for (int kk = 0; kk < K; kk += 32) {
    v16bf bf = load_frag(W, K, n0 + r16, kk, half);
#pragma unroll
    for (int i = 0; i < 4; ++i) {
      v16bf af = load_frag(A, K, m0 + i * 16 + r16, kk, half);
      acc[i] = __builtin_amdgcn_wmma_f32_16x16x32_bf16(false, af, false, bf,
                                                       (short)0, acc[i], false, false);
    }
  }
  const int col = n0 + r16;
  const float bv = bias[col];
#pragma unroll
  for (int i = 0; i < 4; ++i)
#pragma unroll
    for (int r = 0; r < 8; ++r) {
      int row = m0 + i * 16 + half * 8 + r;
      float t = acc[i][r] + bv;
      float ge = 0.5f * t * (1.0f + erff(t * 0.70710678118f));
      out[(size_t)row * N + col] = (bf16)ge;
    }
}

// Dual-accumulate GEMM:
//   mode 0: out = A1@W1^T + alpha*(A2@W2^T + b2)          (k/v/q raw, f32)
//   mode 1: out = sigmoid(A2@W2^T + b2) * (A1@W1^T)       (gated output, f32)
__global__ void gemm_dual_kernel(const bf16* __restrict__ A1, const bf16* __restrict__ W1, int K1,
                                 const bf16* __restrict__ A2, const bf16* __restrict__ W2, int K2,
                                 const float* __restrict__ b2, float alpha, int mode,
                                 float* __restrict__ out, int N) {
  const int wv = threadIdx.x >> 5;
  const int lane = threadIdx.x & 31;
  const int half = lane >> 4;
  const int r16 = lane & 15;
  const int n0 = blockIdx.x * 16;
  const int m0 = blockIdx.y * 512 + wv * 64;

  v8f acc1[4] = {};
  v8f acc2[4] = {};
  for (int kk = 0; kk < K1; kk += 32) {
    v16bf bf = load_frag(W1, K1, n0 + r16, kk, half);
#pragma unroll
    for (int i = 0; i < 4; ++i) {
      v16bf af = load_frag(A1, K1, m0 + i * 16 + r16, kk, half);
      acc1[i] = __builtin_amdgcn_wmma_f32_16x16x32_bf16(false, af, false, bf,
                                                        (short)0, acc1[i], false, false);
    }
  }
  for (int kk = 0; kk < K2; kk += 32) {
    v16bf bf = load_frag(W2, K2, n0 + r16, kk, half);
#pragma unroll
    for (int i = 0; i < 4; ++i) {
      v16bf af = load_frag(A2, K2, m0 + i * 16 + r16, kk, half);
      acc2[i] = __builtin_amdgcn_wmma_f32_16x16x32_bf16(false, af, false, bf,
                                                        (short)0, acc2[i], false, false);
    }
  }
  const int col = n0 + r16;
  const float bv = b2[col];
#pragma unroll
  for (int i = 0; i < 4; ++i)
#pragma unroll
    for (int r = 0; r < 8; ++r) {
      int row = m0 + i * 16 + half * 8 + r;
      float a1 = acc1[i][r];
      float a2 = acc2[i][r] + bv;
      float val = (mode == 0) ? (a1 + alpha * a2)
                              : (a1 / (1.0f + expf(-a2)));
      out[(size_t)row * N + col] = val;
    }
}

// ------------------------------------------------------------ TTT scan
// 48 blocks (b,h), 256 threads: row d = tid>>2 (0..63), quarter qd = tid&3.
// Each thread holds 16 f32 of M's row d (cols qd*16..+15). Dots reduced via
// shfl_xor over the 4 adjacent lanes. out_t read before update (ref order).
__global__ void scan_kernel(const float* __restrict__ kf, const float* __restrict__ vf,
                            const float* __restrict__ qf, const float* __restrict__ lr_scale,
                            bf16* __restrict__ outs, float* __restrict__ Mout) {
  __shared__ float lk[HDd], lv[HDd], lq[HDd];
  const int bh = blockIdx.x;
  const int b = bh / Hh, h = bh % Hh;
  const int d = threadIdx.x >> 2;
  const int qd = threadIdx.x & 3;
  const float lr = 0.2f / (1.0f + expf(-lr_scale[h]));

  float m[16];
#pragma unroll
  for (int j = 0; j < 16; ++j) m[j] = 0.f;

  const size_t base0 = ((size_t)b * Ss) * Pp + (size_t)h * HDd;
  for (int t = 0; t < Ss; ++t) {
    const size_t base = base0 + (size_t)t * Pp;
    const int tid = threadIdx.x;
    if (tid < 64)        lk[tid]       = kf[base + tid];
    else if (tid < 128)  lv[tid - 64]  = vf[base + tid - 64];
    else if (tid < 192)  lq[tid - 128] = qf[base + tid - 128];
    __syncthreads();

    float o = 0.f, p = 0.f, ss = 0.f;
#pragma unroll
    for (int j = 0; j < 16; ++j) {
      int c = qd * 16 + j;
      float kc = lk[c];
      ss += kc * kc;
      o  += m[j] * lq[c];
      p  += m[j] * kc;
    }
    o  += __shfl_xor(o, 1, 32);  o  += __shfl_xor(o, 2, 32);
    p  += __shfl_xor(p, 1, 32);  p  += __shfl_xor(p, 2, 32);
    ss += __shfl_xor(ss, 1, 32); ss += __shfl_xor(ss, 2, 32);

    float rinv = 1.0f / (sqrtf(ss) + 1e-6f);
    float pred = p * rinv;                       // M @ kn
    float coef = (pred + lr * (pred - lv[d])) * rinv;
#pragma unroll
    for (int j = 0; j < 16; ++j) m[j] -= coef * lk[qd * 16 + j];

    if (qd == 0) outs[base + d] = (bf16)o;
    __syncthreads();
  }
  float* mp = Mout + (((size_t)bh * HDd) + d) * HDd + qd * 16;
#pragma unroll
  for (int j = 0; j < 16; ++j) mp[j] = m[j];
}

// ---------------------------------------------------------------- launch

extern "C" void kernel_launch(void* const* d_in, const int* in_sizes, int n_in,
                              void* d_out, int out_size, void* d_ws, size_t ws_size,
                              hipStream_t stream) {
  const float* x      = (const float*)d_in[0];
  const float* ln_g   = (const float*)d_in[1];
  const float* ln_b   = (const float*)d_in[2];
  const float* Wk     = (const float*)d_in[3];
  const float* Wv     = (const float*)d_in[4];
  const float* Wq     = (const float*)d_in[5];
  const float* km_w1  = (const float*)d_in[6];
  const float* km_b1  = (const float*)d_in[7];
  const float* km_w2  = (const float*)d_in[8];
  const float* km_b2  = (const float*)d_in[9];
  const float* vm_w1  = (const float*)d_in[10];
  const float* vm_b1  = (const float*)d_in[11];
  const float* vm_w2  = (const float*)d_in[12];
  const float* vm_b2  = (const float*)d_in[13];
  const float* qm_w1  = (const float*)d_in[14];
  const float* qm_b1  = (const float*)d_in[15];
  const float* qm_w2  = (const float*)d_in[16];
  const float* qm_b2  = (const float*)d_in[17];
  const float* lnk_g  = (const float*)d_in[18];
  const float* lnk_b  = (const float*)d_in[19];
  const float* lnv_g  = (const float*)d_in[20];
  const float* lnv_b  = (const float*)d_in[21];
  const float* lnq_g  = (const float*)d_in[22];
  const float* lnq_b  = (const float*)d_in[23];
  const float* Wo     = (const float*)d_in[24];
  const float* lr_sc  = (const float*)d_in[25];
  const float* Wg     = (const float*)d_in[26];
  const float* bg     = (const float*)d_in[27];

  char* wp = (char*)d_ws;
  auto carve = [&](size_t bytes) -> char* {
    char* p = wp; wp += (bytes + 255) & ~(size_t)255; return p;
  };
  bf16* xn_bf  = (bf16*)carve((size_t)ROWS * Dd * 2);
  bf16* x_bf   = (bf16*)carve((size_t)ROWS * Dd * 2);
  bf16* outs_bf= (bf16*)carve((size_t)ROWS * Dd * 2);
  bf16* Wk_bf  = (bf16*)carve((size_t)Dd * Dd * 2);
  bf16* Wv_bf  = (bf16*)carve((size_t)Dd * Dd * 2);
  bf16* Wq_bf  = (bf16*)carve((size_t)Dd * Dd * 2);
  bf16* Wo_bf  = (bf16*)carve((size_t)Dd * Dd * 2);
  bf16* Wg_bf  = (bf16*)carve((size_t)Dd * Dd * 2);
  bf16* kw1_bf = (bf16*)carve((size_t)DH * Dd * 2);
  bf16* vw1_bf = (bf16*)carve((size_t)DH * Dd * 2);
  bf16* qw1_bf = (bf16*)carve((size_t)DH * Dd * 2);
  bf16* kw2_bf = (bf16*)carve((size_t)Dd * DH * 2);
  bf16* vw2_bf = (bf16*)carve((size_t)Dd * DH * 2);
  bf16* qw2_bf = (bf16*)carve((size_t)Dd * DH * 2);
  bf16* hk_bf  = (bf16*)carve((size_t)ROWS * DH * 2);
  bf16* hv_bf  = (bf16*)carve((size_t)ROWS * DH * 2);
  bf16* hq_bf  = (bf16*)carve((size_t)ROWS * DH * 2);
  float* kraw  = (float*)carve((size_t)ROWS * Dd * 4);
  float* vraw  = (float*)carve((size_t)ROWS * Dd * 4);
  float* qraw  = (float*)carve((size_t)ROWS * Dd * 4);

  auto cvt = [&](const float* src, bf16* dst, int n) {
    cvt_bf16_kernel<<<(n + 255) / 256, 256, 0, stream>>>(src, dst, n);
  };
  cvt(x, x_bf, ROWS * Dd);
  cvt(Wk, Wk_bf, Dd * Dd);  cvt(Wv, Wv_bf, Dd * Dd);  cvt(Wq, Wq_bf, Dd * Dd);
  cvt(Wo, Wo_bf, Dd * Dd);  cvt(Wg, Wg_bf, Dd * Dd);
  cvt(km_w1, kw1_bf, DH * Dd); cvt(vm_w1, vw1_bf, DH * Dd); cvt(qm_w1, qw1_bf, DH * Dd);
  cvt(km_w2, kw2_bf, Dd * DH); cvt(vm_w2, vw2_bf, Dd * DH); cvt(qm_w2, qw2_bf, Dd * DH);

  ln_in_kernel<<<ROWS, 256, 0, stream>>>(x, ln_g, ln_b, xn_bf);

  dim3 blk(256);
  dim3 grid_h(DH / 16, ROWS / 512);   // 12 x 16
  gemm_gelu_kernel<<<grid_h, blk, 0, stream>>>(xn_bf, kw1_bf, km_b1, hk_bf, DH, Dd);
  gemm_gelu_kernel<<<grid_h, blk, 0, stream>>>(xn_bf, vw1_bf, vm_b1, hv_bf, DH, Dd);
  gemm_gelu_kernel<<<grid_h, blk, 0, stream>>>(xn_bf, qw1_bf, qm_b1, hq_bf, DH, Dd);

  dim3 grid_p(Dd / 16, ROWS / 512);   // 48 x 16
  gemm_dual_kernel<<<grid_p, blk, 0, stream>>>(xn_bf, Wk_bf, Dd, hk_bf, kw2_bf, DH,
                                               km_b2, 0.1f, 0, kraw, Dd);
  gemm_dual_kernel<<<grid_p, blk, 0, stream>>>(xn_bf, Wv_bf, Dd, hv_bf, vw2_bf, DH,
                                               vm_b2, 0.1f, 0, vraw, Dd);
  gemm_dual_kernel<<<grid_p, blk, 0, stream>>>(xn_bf, Wq_bf, Dd, hq_bf, qw2_bf, DH,
                                               qm_b2, 0.1f, 0, qraw, Dd);

  const int hl_rows = ROWS * Hh;      // 98304 rows of 64
  head_ln_kernel<<<hl_rows / 8, 256, 0, stream>>>(kraw, lnk_g, lnk_b);
  head_ln_kernel<<<hl_rows / 8, 256, 0, stream>>>(vraw, lnv_g, lnv_b);
  head_ln_kernel<<<hl_rows / 8, 256, 0, stream>>>(qraw, lnq_g, lnq_b);

  float* out_f = (float*)d_out;
  float* M_out = out_f + (size_t)ROWS * Dd;
  scan_kernel<<<Bb * Hh, 256, 0, stream>>>(kraw, vraw, qraw, lr_sc, outs_bf, M_out);

  gemm_dual_kernel<<<grid_p, blk, 0, stream>>>(outs_bf, Wo_bf, Dd, x_bf, Wg_bf, Dd,
                                               bg, 1.0f, 1, out_f, Dd);
}
